// ParamInterpCNN_10995116277926
// MI455X (gfx1250) — matrix operand order repaired and checked by reading, not compiled, not run on hardware
//
#include <hip/hip_runtime.h>
#include <hip/hip_bf16.h>
#include <stdint.h>

// ---------------------------------------------------------------------------
// ParamInterpCNN on MI455X (gfx1250): implicit-GEMM 5x5 convs on WMMA f16,
// with weight slabs streamed into double-buffered LDS by the Tensor Data
// Mover (tensor_load_to_lds + TENSORcnt), overlapped with WMMA compute.
//   act0: rotated input, f16, channels-last, padded [256][68][68][32]
//   act1: conv0+ReLU out, f16, padded [256][68][68][64]
//   act2: conv1+ReLU out, f16, padded [256][68][68][64]
//   out : conv2, f32 [256][64][64]
// ---------------------------------------------------------------------------

typedef __attribute__((ext_vector_type(16))) _Float16 v16h;
typedef __attribute__((ext_vector_type(8)))  _Float16 v8h;
typedef __attribute__((ext_vector_type(8)))  float    v8f;
typedef __attribute__((ext_vector_type(4)))  unsigned int u32x4;
typedef __attribute__((ext_vector_type(8)))  unsigned int u32x8;

#define NB   256
#define NC   32
#define NH   64
#define NW   64
#define PAD  2
#define PDIM 68   // 64 + 2*PAD

// ---------------------------------------------------------------------------
// Weight prep: OIHW f32 -> [oc][(ky*5+kx)*CIN + ci] f16 (K contiguous per oc)
// and w2 (1,64,5,5) -> w2p[(ky*5+kx)*64 + c] f32
// ---------------------------------------------------------------------------
__global__ __launch_bounds__(256) void prep_weights(const float* __restrict__ w0,
                                                    const float* __restrict__ w1,
                                                    const float* __restrict__ w2,
                                                    _Float16* __restrict__ wB0,
                                                    _Float16* __restrict__ wB1,
                                                    float* __restrict__ w2p) {
  int i = blockIdx.x * 256 + threadIdx.x;
  if (i < 64 * 32 * 25) {            // w0: [64][32][25] flat
    int t  = i % 25;
    int c  = (i / 25) % 32;
    int oc = i / (25 * 32);
    wB0[(size_t)oc * 800 + t * 32 + c] = (_Float16)w0[i];
  }
  if (i < 64 * 64 * 25) {            // w1: [64][64][25] flat
    int t  = i % 25;
    int c  = (i / 25) % 64;
    int oc = i / (25 * 64);
    wB1[(size_t)oc * 1600 + t * 64 + c] = (_Float16)w1[i];
  }
  if (i < 64 * 25) {                 // w2: [64][25] flat
    int t = i % 25;
    int c = i / 25;
    w2p[t * 64 + c] = w2[i];
  }
}

// ---------------------------------------------------------------------------
// Rotate + expand. Pre-rotation channel image is constant along y, so each
// rotated pixel = (y-validity weight) * 1-D bilinear sample of x[b,c,:].
// Writes full padded buffer (borders = 0) with 16B vector stores.
// ---------------------------------------------------------------------------
__global__ __launch_bounds__(256) void rotate_expand(const float* __restrict__ x,
                                                     _Float16* __restrict__ act0) {
  int idx = blockIdx.x * 256 + threadIdx.x;        // b*68*68 + yy*68 + xx
  if (idx >= NB * PDIM * PDIM) return;
  int xx = idx % PDIM;
  int t  = idx / PDIM;
  int yy = t % PDIM;
  int b  = t / PDIM;

  union { v8h v[4]; _Float16 h[NC]; } u;
  v8h* out = (v8h*)(act0 + (size_t)idx * NC);      // 64B-aligned

  if (yy < PAD || yy >= PDIM - PAD || xx < PAD || xx >= PDIM - PAD) {
    v8h z = {};
#pragma unroll
    for (int i = 0; i < 4; ++i) out[i] = z;
    return;
  }
  const float cx = 31.5f;
  float dx = (float)(xx - PAD) - cx;
  float dy = (float)(yy - PAD) - cx;
  const float* xrow = x + (size_t)b * NC * NW;     // x: [B][1][C][W]
  for (int c = 0; c < NC; ++c) {
    float ang = (-3.14159265358979323846f / 32.0f) * (float)c;
    float ca = __cosf(ang), sa = __sinf(ang);
    float xs = ca * dx + sa * dy + cx;
    float ys = -sa * dx + ca * dy + cx;
    float x0f = floorf(xs), y0f = floorf(ys);
    float wx1 = xs - x0f, wy1 = ys - y0f;
    int x0 = (int)x0f, y0 = (int)y0f;
    int x1 = x0 + 1,  y1 = y0 + 1;
    float vy = ((y0 >= 0 && y0 < NH) ? (1.0f - wy1) : 0.0f) +
               ((y1 >= 0 && y1 < NH) ? wy1 : 0.0f);
    float v0 = (x0 >= 0 && x0 < NW) ? xrow[c * NW + x0] * (1.0f - wx1) : 0.0f;
    float v1 = (x1 >= 0 && x1 < NW) ? xrow[c * NW + x1] * wx1 : 0.0f;
    u.h[c] = (_Float16)((v0 + v1) * vy);
  }
#pragma unroll
  for (int i = 0; i < 4; ++i) out[i] = u.v[i];
}

// ---------------------------------------------------------------------------
// Zero fill (for padded borders of act1/act2; conv kernels rewrite interiors)
// ---------------------------------------------------------------------------
__global__ __launch_bounds__(256) void zero_f4(float4* __restrict__ p, size_t n) {
  size_t i      = (size_t)blockIdx.x * blockDim.x + threadIdx.x;
  size_t stride = (size_t)gridDim.x * blockDim.x;
  float4 z; z.x = 0.f; z.y = 0.f; z.z = 0.f; z.w = 0.f;
  for (; i < n; i += stride) p[i] = z;
}

// ---------------------------------------------------------------------------
// Fragment helper: two contiguous 16B loads -> v16h
// ---------------------------------------------------------------------------
__device__ __forceinline__ v16h load_frag16(const _Float16* p0, const _Float16* p1) {
  union { v16h v; v8h h[2]; } u;
  u.h[0] = *(const v8h*)p0;
  u.h[1] = *(const v8h*)p1;
  return u.v;
}

// ---------------------------------------------------------------------------
// TDM: DMA a 2D weight tile (64 oc rows x CIN halves, row stride KTOT halves)
// from global into LDS, with LDS row padding of +8 halves via D# pad fields.
// D# group layouts per CDNA5 ISA 8.3/8.4. 2D tensor -> VADDR2/3 = NULL.
// ---------------------------------------------------------------------------
__device__ __forceinline__ void tdm_load_weights(u32x4 g0, u32x8 g1) {
  asm volatile("tensor_load_to_lds %0, %1" :: "s"(g0), "s"(g1) : "memory");
}

// ---------------------------------------------------------------------------
// Implicit-GEMM 5x5 SAME conv, CIN -> 64, bias + ReLU, f16 in/out (padded).
// One wave per (b, y, x-tile of 32), all 64 output channels.
// Per 32-K chunk: 4 A global_load_b128 + 4 B ds_load_b128 pairs -> 8 WMMAs.
// Weights arrive via TDM into double-buffered LDS; wave 0 drives the DMA
// pipeline (TENSORcnt is in-order, so wait<=1 means current slab is ready).
// Grid is exactly NB*NH*2 wave-tiles so EXEC is all ones for WMMA.
// ---------------------------------------------------------------------------
template <int CIN>
__global__ __launch_bounds__(256) void conv_wmma(const _Float16* __restrict__ actin,
                                                 const _Float16* __restrict__ wB,
                                                 const float* __restrict__ bias,
                                                 _Float16* __restrict__ actout) {
  constexpr int LROW    = CIN + 8;               // padded LDS row (halves)
  constexpr int BUFH    = 64 * LROW;             // halves per slab buffer
  __shared__ _Float16 lw[2 * BUFH];

  const int tid  = threadIdx.x;
  const int lane = tid & 31;
  const int wave = tid >> 5;
  const int wid  = blockIdx.x * 8 + wave;        // 0 .. 32767
  const int b    = wid >> 7;
  const int rem  = wid & 127;
  const int y    = rem >> 1;                     // output row 0..63
  const int x0   = (rem & 1) << 5;               // output col base 0 or 32

  const int m    = lane & 15;                    // A row / B,N col / D col
  const int hi   = lane >> 4;                    // lane-half select
  const int kb   = hi * 8;                       // A K sub-base (0 or 8)
  const int KTOT = 25 * CIN;

  // ---- TDM descriptor (uniform scalars) ----
  const unsigned ldsBase  = (unsigned)(uintptr_t)(&lw[0]);   // LDS byte offset
  const unsigned bufBytes = (unsigned)(BUFH * 2);
  const uint64_t gBase    = (uint64_t)(uintptr_t)wB;
  // group1: workgroup_mask=0, data_size=1(2B), pad_enable, pad_interval:
  //   CIN=32 row=64B  -> code 3 (8x8B);  CIN=64 row=128B -> code 4 (16x8B)
  // pad_amount code 3 -> 4 DWORDs = 16B = 8 halves of row padding.
  u32x8 g1;
  g1[0] = 0x00010000u | (1u << 20) | ((CIN == 32 ? 3u : 4u) << 22) | (3u << 25);
  g1[1] = ((unsigned)KTOT & 0xFFFFu) << 16;        // tensor_dim0 lo16 (hi16=0)
  g1[2] = ((unsigned)KTOT >> 16) | (64u << 16);    // tensor_dim0 hi | tensor_dim1 lo
  g1[3] = ((unsigned)CIN) << 16;                   // tensor_dim1 hi=0 | tile_dim0
  g1[4] = 64u;                                     // tile_dim1=64, tile_dim2=0
  g1[5] = (unsigned)KTOT;                          // tensor_dim0_stride lo32
  g1[6] = 0u;                                      // stride hi | dim1_stride lo
  g1[7] = 0u;

  v8f acc[2][4];
  {
    v8f z = {};
#pragma unroll
    for (int s = 0; s < 2; ++s)
#pragma unroll
      for (int j = 0; j < 4; ++j) acc[s][j] = z;
  }

  // prologue: DMA tap 0 slab into buffer 0
  if (wave == 0) {
    uint64_t ga = gBase;                           // tap 0
    u32x4 g0;
    g0[0] = 1u;                                    // count=1 (valid descriptor)
    g0[1] = ldsBase;                               // lds_addr
    g0[2] = (unsigned)ga;                          // global_addr[31:0]
    g0[3] = (unsigned)(ga >> 32) | (2u << 30);     // global_addr[56:32] | type=2
    tdm_load_weights(g0, g1);
  }

  int ky = 0, kx = 0;
  for (int t = 0; t < 25; ++t) {
    __syncthreads();   // everyone done reading the buffer TDM will next fill
    if (wave == 0) {
      if (t < 24) {
        uint64_t ga = gBase + (uint64_t)(t + 1) * (CIN * 2);
        u32x4 g0;
        g0[0] = 1u;
        g0[1] = ldsBase + (unsigned)((t + 1) & 1) * bufBytes;
        g0[2] = (unsigned)ga;
        g0[3] = (unsigned)(ga >> 32) | (2u << 30);
        tdm_load_weights(g0, g1);
        __builtin_amdgcn_s_wait_tensorcnt(1);      // tap t slab complete
      } else {
        __builtin_amdgcn_s_wait_tensorcnt(0);
      }
    }
    __syncthreads();   // slab for tap t visible to all waves
    const _Float16* lwb = &lw[(t & 1) * BUFH];

    // A rows: pixels (y+ky, x0+kx+m) and (y+ky, x0+16+kx+m), channels-last
    const _Float16* ap0 =
        actin + ((((size_t)b * PDIM) + (size_t)(y + ky)) * PDIM + (size_t)(x0 + kx + m)) * CIN;
    const _Float16* ap1 = ap0 + (size_t)16 * CIN;

    if (t < 24) {  // prefetch next tap's A rows (uniform branch, EXEC intact)
      int nky = ky, nkx = kx + 1;
      if (nkx == 5) { nkx = 0; ++nky; }
      const _Float16* np =
          actin + ((((size_t)b * PDIM) + (size_t)(y + nky)) * PDIM + (size_t)(x0 + nkx + m)) * CIN;
      __builtin_prefetch(np, 0, 1);
      __builtin_prefetch(np + 16 * CIN, 0, 1);
    }

#pragma unroll
    for (int h = 0; h < CIN / 32; ++h) {
      // A 16x32 f16 fragment layout: lane<16 K={0..7,16..23}; lane>=16 K={8..15,24..31}
      v16h a0 = load_frag16(ap0 + h * 32 + kb, ap0 + h * 32 + 16 + kb);
      v16h a1 = load_frag16(ap1 + h * 32 + kb, ap1 + h * 32 + 16 + kb);
      const int kl = h * 32 + kb;                // B: same K split as A
#pragma unroll
      for (int j = 0; j < 4; ++j) {
        const _Float16* wp = &lwb[(j * 16 + m) * LROW + kl];
        v16h bf = load_frag16(wp, wp + 16);
        acc[0][j] = __builtin_amdgcn_wmma_f32_16x16x32_f16(false, a0, false, bf, (short)0,
                                                           acc[0][j], false, false);
        acc[1][j] = __builtin_amdgcn_wmma_f32_16x16x32_f16(false, a1, false, bf, (short)0,
                                                           acc[1][j], false, false);
      }
    }
    if (++kx == 5) { kx = 0; ++ky; }
  }

  // Epilogue: D element acc[s][j][v] = D[M = v + 8*hi][N = m]; oc = j*16 + m
  const float bb0 = bias[m];
  const float bb1 = bias[16 + m];
  const float bb2 = bias[32 + m];
  const float bb3 = bias[48 + m];
  const size_t rowBase = (((size_t)b * PDIM) + (size_t)(y + PAD)) * PDIM;
#pragma unroll
  for (int s = 0; s < 2; ++s) {
#pragma unroll
    for (int v = 0; v < 8; ++v) {
      const int X = x0 + s * 16 + v + 8 * hi + PAD;          // padded x
      const size_t base = (rowBase + (size_t)X) * 64 + (size_t)m;
      actout[base]      = (_Float16)fmaxf(acc[s][0][v] + bb0, 0.0f);
      actout[base + 16] = (_Float16)fmaxf(acc[s][1][v] + bb1, 0.0f);
      actout[base + 32] = (_Float16)fmaxf(acc[s][2][v] + bb2, 0.0f);
      actout[base + 48] = (_Float16)fmaxf(acc[s][3][v] + bb3, 0.0f);
    }
  }
}

// ---------------------------------------------------------------------------
// conv2: 64 -> 1, no ReLU, f32 accumulate; ~1% of total FLOPs
// ---------------------------------------------------------------------------
__global__ __launch_bounds__(256) void conv_last(const _Float16* __restrict__ act2,
                                                 const float* __restrict__ w2p,
                                                 const float* __restrict__ b2,
                                                 float* __restrict__ out) {
  int idx = blockIdx.x * 256 + threadIdx.x;     // b*4096 + y*64 + x
  if (idx >= NB * NH * NW) return;
  int x = idx & 63;
  int y = (idx >> 6) & 63;
  int b = idx >> 12;
  float s = b2[0];
  int t = 0;
  for (int ky = 0; ky < 5; ++ky) {
    for (int kx = 0; kx < 5; ++kx) {
      const v8h* p = (const v8h*)(act2 +
          ((((size_t)b * PDIM) + (size_t)(y + ky)) * PDIM + (size_t)(x + kx)) * 64);
      const float* wp = w2p + t * 64;           // uniform across lanes -> scalar loads
#pragma unroll
      for (int i = 0; i < 8; ++i) {
        v8h q = p[i];
#pragma unroll
        for (int e = 0; e < 8; ++e) s += (float)q[e] * wp[i * 8 + e];
      }
      ++t;
    }
  }
  out[idx] = s;
}

// ---------------------------------------------------------------------------
extern "C" void kernel_launch(void* const* d_in, const int* in_sizes, int n_in,
                              void* d_out, int out_size, void* d_ws, size_t ws_size,
                              hipStream_t stream) {
  const float* x  = (const float*)d_in[0];
  const float* w0 = (const float*)d_in[1];
  const float* b0 = (const float*)d_in[2];
  const float* w1 = (const float*)d_in[3];
  const float* b1 = (const float*)d_in[4];
  const float* w2 = (const float*)d_in[5];
  const float* b2 = (const float*)d_in[6];
  float* out = (float*)d_out;

  char* ws = (char*)d_ws;
  // workspace layout (all 256B aligned)
  const size_t WB0_B  = 64 * 800 * 2;                        // 102400
  const size_t WB1_B  = 64 * 1600 * 2;                       // 204800
  const size_t W2P_B  = 1600 * 4;                            //   6400
  const size_t ACT0_B = (size_t)NB * PDIM * PDIM * 32 * 2;   //  75,759,616
  const size_t ACT1_B = (size_t)NB * PDIM * PDIM * 64 * 2;   // 151,519,232
  _Float16* wB0  = (_Float16*)(ws);
  _Float16* wB1  = (_Float16*)(ws + WB0_B);
  float*    w2p  = (float*)(ws + WB0_B + WB1_B);
  _Float16* act0 = (_Float16*)(ws + WB0_B + WB1_B + W2P_B);
  _Float16* act1 = (_Float16*)(ws + WB0_B + WB1_B + W2P_B + ACT0_B);
  _Float16* act2 = (_Float16*)(ws + WB0_B + WB1_B + W2P_B + ACT0_B + ACT1_B);

  prep_weights<<<(64 * 64 * 25 + 255) / 256, 256, 0, stream>>>(w0, w1, w2, wB0, wB1, w2p);
  rotate_expand<<<(NB * PDIM * PDIM + 255) / 256, 256, 0, stream>>>(x, act0);
  // zero act1+act2 (contiguous) so padded borders are 0 before interiors written
  zero_f4<<<4096, 256, 0, stream>>>((float4*)act1, (2 * ACT1_B) / 16);
  conv_wmma<32><<<4096, 256, 0, stream>>>(act0, wB0, b0, act1);
  conv_wmma<64><<<4096, 256, 0, stream>>>(act1, wB1, b1, act2);
  conv_last<<<(NB * NH * NW + 255) / 256, 256, 0, stream>>>(act2, w2p, b2, out);
}
